// MyModel_40742059769996
// MI455X (gfx1250) — matrix operand order repaired and checked by reading, not compiled, not run on hardware
//
#include <hip/hip_runtime.h>
#include <hip/hip_bf16.h>

typedef _Float16 v8h  __attribute__((ext_vector_type(8)));
typedef _Float16 v16h __attribute__((ext_vector_type(16)));
typedef float    v8f  __attribute__((ext_vector_type(8)));

#define GM 3072
#define GK 2048
#define GN 2048
#define NIMG 3072            // CC*STD
#define HALFELEMS 6291456ULL // 3072*64*32
#define EPS 1e-3f

// ---------------------------------------------------------------------------
// Fold BN scale into W and transpose: Wt[n][k] = (g[k]*rsqrt(v[k]+eps)) * W[k][n]  (f16)
// block (32,8), grid (64,64)
// ---------------------------------------------------------------------------
__global__ void prep_wt(const float* __restrict__ W, const float* __restrict__ g,
                        const float* __restrict__ vv, _Float16* __restrict__ Wt)
{
    __shared__ float tile[32][33];
    const int nb = blockIdx.x * 32;
    const int kb = blockIdx.y * 32;
    for (int i = threadIdx.y; i < 32; i += 8) {
        const int k = kb + i;
        const float sc = g[k] * rsqrtf(vv[k] + EPS);
        tile[i][threadIdx.x] = W[(size_t)k * GN + nb + threadIdx.x] * sc;
    }
    __syncthreads();
    for (int i = threadIdx.y; i < 32; i += 8) {
        // write Wt[nb+i][kb+tx] : coalesced in k
        Wt[(size_t)(nb + i) * GK + kb + threadIdx.x] = (_Float16)tile[threadIdx.x][i];
    }
}

// ---------------------------------------------------------------------------
// Folded bias: b'[n] = b[n] + sum_k (be[k] - m[k]*scale[k]) * W[k][n]
// grid 8, block 256 (coalesced row sweeps)
// ---------------------------------------------------------------------------
__global__ void prep_bias(const float* __restrict__ W, const float* __restrict__ g,
                          const float* __restrict__ be, const float* __restrict__ mm,
                          const float* __restrict__ vv, const float* __restrict__ b,
                          float* __restrict__ biasp)
{
    const int n = blockIdx.x * 256 + threadIdx.x;
    float acc = b[n];
    for (int k = 0; k < GK; ++k) {
        const float sc = g[k] * rsqrtf(vv[k] + EPS);
        const float sh = be[k] - mm[k] * sc;
        acc += sh * W[(size_t)k * GN + n];
    }
    biasp[n] = acc;
}

// ---------------------------------------------------------------------------
// Checkerboard split of x; also copy x -> out0 and x[:,1:] -> out5
// one thread per (n,r,c) in [3072,64,32]
// ---------------------------------------------------------------------------
__global__ void split_x(const float* __restrict__ x, float* __restrict__ u1,
                        float* __restrict__ u2, _Float16* __restrict__ u2h,
                        float* __restrict__ outx, float* __restrict__ outxs)
{
    const size_t t = (size_t)blockIdx.x * 256 + threadIdx.x;
    if (t >= HALFELEMS) return;
    const int c = (int)(t & 31);
    const int r = (int)((t >> 5) & 63);
    const size_t n = t >> 11;
    const size_t base = (n << 12) + (size_t)r * 64 + 2 * c;
    const float xe = x[base], xo = x[base + 1];
    const float a  = (r & 1) ? xo : xe;   // u1: masked half
    const float bb = (r & 1) ? xe : xo;   // u2: demasked half
    u1[t] = a;
    u2[t] = bb;
    u2h[t] = (_Float16)bb;
    outx[base] = xe; outx[base + 1] = xo;
    const size_t cci = n / 12, s = n % 12;
    if (s >= 1) {
        const size_t b2 = ((cci * 11 + (s - 1)) << 12) + (size_t)r * 64 + 2 * c;
        outxs[b2] = xe; outxs[b2 + 1] = xo;
    }
}

// ---------------------------------------------------------------------------
// Plain checkerboard split (decoder inputs): y -> w1 (f32+f16), w2 (f32)
// ---------------------------------------------------------------------------
__global__ void split_plain(const float* __restrict__ y, float* __restrict__ w1,
                            _Float16* __restrict__ w1h, float* __restrict__ w2)
{
    const size_t t = (size_t)blockIdx.x * 256 + threadIdx.x;
    if (t >= HALFELEMS) return;
    const int c = (int)(t & 31);
    const int r = (int)((t >> 5) & 63);
    const size_t n = t >> 11;
    const size_t base = (n << 12) + (size_t)r * 64 + 2 * c;
    const float xe = y[base], xo = y[base + 1];
    const float a  = (r & 1) ? xo : xe;
    const float bb = (r & 1) ? xe : xo;
    w1[t] = a;
    w1h[t] = (_Float16)a;
    w2[t] = bb;
}

// ---------------------------------------------------------------------------
// _mix: even rows: y[2c]=w1, y[2c+1]=w2 ; odd rows: y[(2c+63)&63]=w1, y[(2c+2)&63]=w2
// optionally also emit [:,1:] sub-copy (for phiX)
// ---------------------------------------------------------------------------
__global__ void mix_kernel(const float* __restrict__ w1, const float* __restrict__ w2,
                           float* __restrict__ y, float* __restrict__ ysub)
{
    const size_t t = (size_t)blockIdx.x * 256 + threadIdx.x;
    if (t >= HALFELEMS) return;
    const int c = (int)(t & 31);
    const int r = (int)((t >> 5) & 63);
    const size_t n = t >> 11;
    const float a = w1[t], bb = w2[t];
    int j1, j2;
    if ((r & 1) == 0) { j1 = 2 * c; j2 = 2 * c + 1; }
    else              { j1 = (2 * c + 63) & 63; j2 = (2 * c + 2) & 63; }
    const size_t base = (n << 12) + (size_t)r * 64;
    y[base + j1] = a;
    y[base + j2] = bb;
    if (ysub) {
        const size_t cci = n / 12, s = n % 12;
        if (s >= 1) {
            const size_t b2 = ((cci * 11 + (s - 1)) << 12) + (size_t)r * 64;
            ysub[b2 + j1] = a;
            ysub[b2 + j2] = bb;
        }
    }
}

// ---------------------------------------------------------------------------
// Koopman: out[cci',s,r,c] = sum_k phi[(cci'%4)*64+cci'/4, s, r, k] * kop[k,c]
// block = 256 (4 rows x 64 cols); grid = 3072*16
// ---------------------------------------------------------------------------
__global__ void koopman(const float* __restrict__ phi, const float* __restrict__ kop,
                        float* __restrict__ out)
{
    __shared__ float kS[64 * 64];
    __shared__ float rS[4 * 64];
    for (int i = threadIdx.x; i < 4096; i += 256) kS[i] = kop[i];
    const int bid = blockIdx.x;
    const int oi = bid >> 4;                 // output image in [0,3072)
    const int rg = bid & 15;                 // 4-row group
    const int cci2 = oi / 12, s = oi % 12;
    const int cci = (cci2 & 3) * 64 + (cci2 >> 2);   // batch permutation
    const size_t inBase = ((size_t)(cci * 12 + s) << 12) + (size_t)rg * 256;
    rS[threadIdx.x] = phi[inBase + threadIdx.x];
    __syncthreads();
    const int lr = threadIdx.x >> 6;
    const int c = threadIdx.x & 63;
    float acc = 0.f;
#pragma unroll
    for (int k = 0; k < 64; ++k) acc += rS[lr * 64 + k] * kS[k * 64 + c];
    out[((size_t)oi << 12) + (size_t)rg * 256 + threadIdx.x] = acc;
}

// ---------------------------------------------------------------------------
// WMMA GEMM: out = resid + sign * (A @ Wt^T + bias)
// A: [3072 x 2048] f16 row-major (activations)
// Wt: [2048 x 2048] f16, row n holds W'[.,n] contiguous in k  (pre-transposed)
// Block: 256 threads = 8 waves (2x4), wave tile 64x64, block tile 128x256.
// grid (24, 8)
// ---------------------------------------------------------------------------
__launch_bounds__(256, 1)
__global__ void gemm_wmma(const _Float16* __restrict__ A,
                          const _Float16* __restrict__ Wt,
                          const float* __restrict__ bias,
                          const float* __restrict__ resid,
                          float sign,
                          float* __restrict__ outF,
                          _Float16* __restrict__ outH)
{
    const int lane = threadIdx.x & 31;
    const int wave = threadIdx.x >> 5;
    const int wm = wave & 1;        // wave M position (0..1)
    const int wn = wave >> 1;       // wave N position (0..3)
    const int l16 = lane & 15;
    const int lh  = lane >> 4;      // lane half
    const int row0 = blockIdx.x * 128 + wm * 64;
    const int col0 = blockIdx.y * 256 + wn * 64;

    v8f acc[4][4] = {};

    for (int kb = 0; kb < GK; kb += 32) {
        v16h a[4], b[4];
#pragma unroll
        for (int i = 0; i < 4; ++i) {
            // A 16x32 f16 fragment: lanes 0-15 hold K 0..15, lanes 16-31 K 16..31
            const _Float16* ap = A + (size_t)(row0 + i * 16 + l16) * GK + kb + lh * 16;
            const v8h lo = *(const v8h*)ap;
            const v8h hi = *(const v8h*)(ap + 8);
            a[i] = __builtin_shufflevector(lo, hi, 0, 1, 2, 3, 4, 5, 6, 7,
                                                   8, 9, 10, 11, 12, 13, 14, 15);
        }
#pragma unroll
        for (int j = 0; j < 4; ++j) {
            // B 32x16 fragment: lane = column n; lanes 0-15 K 0..15, lanes 16-31 K 16..31
            const _Float16* bp = Wt + (size_t)(col0 + j * 16 + l16) * GK + kb + lh * 16;
            b[j] = *(const v16h*)bp;
        }
#pragma unroll
        for (int i = 0; i < 4; ++i)
#pragma unroll
            for (int j = 0; j < 4; ++j)
                acc[i][j] = __builtin_amdgcn_wmma_f32_16x16x32_f16(
                    false, a[i], false, b[j], (short)0, acc[i][j], false, false);
    }

    // Epilogue: C/D layout -> n = lane%16, m = v + 8*(lane/16)
#pragma unroll
    for (int j = 0; j < 4; ++j) {
        const int n = col0 + j * 16 + l16;
        const float bb = bias[n];
#pragma unroll
        for (int i = 0; i < 4; ++i) {
#pragma unroll
            for (int v = 0; v < 8; ++v) {
                const int m = row0 + i * 16 + lh * 8 + v;
                const size_t idx = (size_t)m * GN + n;
                const float val = resid[idx] + sign * (acc[i][j][v] + bb);
                outF[idx] = val;
                if (outH) outH[idx] = (_Float16)val;
            }
        }
    }
}

// ---------------------------------------------------------------------------
extern "C" void kernel_launch(void* const* d_in, const int* in_sizes, int n_in,
                              void* d_out, int out_size, void* d_ws, size_t ws_size,
                              hipStream_t stream)
{
    (void)in_sizes; (void)n_in; (void)out_size; (void)ws_size;
    const float* x   = (const float*)d_in[0];
    // d_in[1] = indexs (unused)
    const float* W1  = (const float*)d_in[2];
    const float* b1  = (const float*)d_in[3];
    const float* g1  = (const float*)d_in[4];
    const float* be1 = (const float*)d_in[5];
    const float* m1  = (const float*)d_in[6];
    const float* v1_ = (const float*)d_in[7];
    const float* W3  = (const float*)d_in[8];
    const float* b3  = (const float*)d_in[9];
    const float* g3  = (const float*)d_in[10];
    const float* be3 = (const float*)d_in[11];
    const float* m3  = (const float*)d_in[12];
    const float* v3_ = (const float*)d_in[13];
    const float* kop = (const float*)d_in[14];

    float* o = (float*)d_out;
    const size_t IMG = 12582912ULL;          // 256*12*64*64
    float* oX   = o;
    float* oPhi = o + IMG;
    float* oDec = o + 2 * IMG;
    float* oK   = o + 3 * IMG;
    float* oDK  = o + 4 * IMG;
    float* oXs  = o + 5 * IMG;               // x[:,1:]
    float* oPs  = oXs + 11534336ULL;         // phiX[:,1:]

    char* w = (char*)d_ws;
    float*    bufA = (float*)w;     w += 25165824;   // [3072,2048] f32
    float*    bufB = (float*)w;     w += 25165824;
    _Float16* hA   = (_Float16*)w;  w += 12582912;   // [3072,2048] f16
    _Float16* hB   = (_Float16*)w;  w += 12582912;
    _Float16* Wt1  = (_Float16*)w;  w += 8388608;    // [2048,2048] f16 transposed
    _Float16* Wt3  = (_Float16*)w;  w += 8388608;
    float*    b1p  = (float*)w;     w += 8192;
    float*    b3p  = (float*)w;     w += 8192;

    const dim3 tb(32, 8);
    prep_wt<<<dim3(64, 64), tb, 0, stream>>>(W1, g1, v1_, Wt1);
    prep_wt<<<dim3(64, 64), tb, 0, stream>>>(W3, g3, v3_, Wt3);
    prep_bias<<<8, 256, 0, stream>>>(W1, g1, be1, m1, v1_, b1, b1p);
    prep_bias<<<8, 256, 0, stream>>>(W3, g3, be3, m3, v3_, b3, b3p);

    // encoder
    split_x<<<24576, 256, 0, stream>>>(x, bufA, bufB, hA, oX, oXs);
    const dim3 gg(24, 8);
    gemm_wmma<<<gg, 256, 0, stream>>>(hA, Wt1, b1p, bufA,  1.f, bufA, hB);            // v1
    gemm_wmma<<<gg, 256, 0, stream>>>(hB, Wt3, b3p, bufB,  1.f, bufB, (_Float16*)0);  // w2
    mix_kernel<<<24576, 256, 0, stream>>>(bufA, bufB, oPhi, oPs);

    // decoder(phiX)
    split_plain<<<24576, 256, 0, stream>>>(oPhi, bufA, hA, bufB);
    gemm_wmma<<<gg, 256, 0, stream>>>(hA, Wt3, b3p, bufB, -1.f, bufB, hB);            // v2
    gemm_wmma<<<gg, 256, 0, stream>>>(hB, Wt1, b1p, bufA, -1.f, bufA, (_Float16*)0);  // u1
    mix_kernel<<<24576, 256, 0, stream>>>(bufA, bufB, oDec, (float*)0);

    // Koopman step
    koopman<<<49152, 256, 0, stream>>>(oPhi, kop, oK);

    // decoder(kPhix)
    split_plain<<<24576, 256, 0, stream>>>(oK, bufA, hA, bufB);
    gemm_wmma<<<gg, 256, 0, stream>>>(hA, Wt3, b3p, bufB, -1.f, bufB, hB);
    gemm_wmma<<<gg, 256, 0, stream>>>(hB, Wt1, b1p, bufA, -1.f, bufA, (_Float16*)0);
    mix_kernel<<<24576, 256, 0, stream>>>(bufA, bufB, oDK, (float*)0);
}
